// _Expert_644245095186
// MI455X (gfx1250) — matrix operand order repaired and checked by reading, not compiled, not run on hardware
//
#include <hip/hip_runtime.h>

// ---------------------------------------------------------------------------
// MoE SwiGLU expert FFN for MI455X (gfx1250), wave32 + WMMA bf16.
//   E=8 experts, 1024 tokens/expert, D=2048, H=5632, fp32 in/out.
// Kernel 1: h' = silu(x*w1^T) * (x*w3^T)   (fused, shared A tiles) -> bf16 ws
// Kernel 2: out = h' * w2^T                -> fp32 d_out
//
// v_wmma_f32_16x16x32_bf16 math (fp32 accumulate). fp32 operands converted to
// bf16 while staging global->LDS (per-expert working set fits in 192MB L2).
// LDS double-buffered, one barrier per K-slice. Kernel 2 stages its bf16 A
// tile with gfx1250 async copies (global_load_async_to_lds_b128 + ASYNCcnt).
// ---------------------------------------------------------------------------

typedef __attribute__((ext_vector_type(16))) __bf16 v16bf;
typedef __attribute__((ext_vector_type(8)))  __bf16 bf16x8;
typedef __attribute__((ext_vector_type(8)))  float  v8f;

constexpr int E   = 8;
constexpr int T   = 8192;
constexpr int D   = 2048;
constexpr int HD  = 5632;
constexpr int TPE = T / E;           // 1024 tokens per expert

constexpr int BM  = 128;             // block tile M (tokens)
constexpr int BN  = 256;             // block tile N
constexpr int BK  = 32;              // K slice == one WMMA K step (bf16)
constexpr int BKP = BK + 8;          // LDS pitch in bf16 (80B) -> conflict-free

// ---- LDS fragment loads (CDNA5 wave32 WMMA layouts) -----------------------
// A 16x32 bf16: lanes 0-15 hold M=lane, K {0..7, 16..23}; lanes 16-31 hold
// M=lane-16, K {8..15, 24..31}. Two 16B chunks per lane.
__device__ inline v16bf load_frag_A(const __bf16* base, int lane) {
  const int r  = lane & 15;
  const int kh = lane >> 4;                 // 0 or 1
  const __bf16* p = base + r * BKP + kh * 8;
  union { v16bf v; bf16x8 h[2]; } u;
  u.h[0] = *(const bf16x8*)(p);
  u.h[1] = *(const bf16x8*)(p + 16);
  return u.v;
}

// B 32x16 bf16 (K x N): lanes 0-15 hold N=lane, K {0..7, 16..23}; lanes 16-31
// hold N=lane-16, K {8..15, 24..31}. B rows stored as [N][K] (w rows, K contiguous).
__device__ inline v16bf load_frag_B(const __bf16* base, int lane) {
  const int n  = lane & 15;
  const int kh = lane >> 4;
  const __bf16* p = base + n * BKP + kh * 8;
  union { v16bf v; bf16x8 h[2]; } u;
  u.h[0] = *(const bf16x8*)(p);
  u.h[1] = *(const bf16x8*)(p + 16);
  return u.v;
}

// ---- cooperative global -> LDS staging ------------------------------------
// 128 rows x 32 cols fp32 tile, 256 threads: 2 threads/row, 16 floats each.
__device__ inline void stage_f32_tile128(__bf16* dst, const float* src, int ld,
                                         int tid) {
  const int row = tid >> 1;
  const int col = (tid & 1) * 16;
  const float4* g = (const float4*)(src + (size_t)row * ld + col);
  float4 f0 = g[0], f1 = g[1], f2 = g[2], f3 = g[3];
  bf16x8 b0, b1;
  b0[0]=(__bf16)f0.x; b0[1]=(__bf16)f0.y; b0[2]=(__bf16)f0.z; b0[3]=(__bf16)f0.w;
  b0[4]=(__bf16)f1.x; b0[5]=(__bf16)f1.y; b0[6]=(__bf16)f1.z; b0[7]=(__bf16)f1.w;
  b1[0]=(__bf16)f2.x; b1[1]=(__bf16)f2.y; b1[2]=(__bf16)f2.z; b1[3]=(__bf16)f2.w;
  b1[4]=(__bf16)f3.x; b1[5]=(__bf16)f3.y; b1[6]=(__bf16)f3.z; b1[7]=(__bf16)f3.w;
  __bf16* d = dst + row * BKP + col;
  *(bf16x8*)(d)     = b0;
  *(bf16x8*)(d + 8) = b1;
}

// 256 rows x 32 cols fp32 tile, 256 threads: 1 thread/row, 32 floats each.
__device__ inline void stage_f32_tile256(__bf16* dst, const float* src, int ld,
                                         int tid) {
  const float4* g = (const float4*)(src + (size_t)tid * ld);
  float4 f[8];
#pragma unroll
  for (int i = 0; i < 8; ++i) f[i] = g[i];
  __bf16* d = dst + tid * BKP;
#pragma unroll
  for (int i = 0; i < 4; ++i) {
    bf16x8 b;
    b[0]=(__bf16)f[2*i].x;   b[1]=(__bf16)f[2*i].y;
    b[2]=(__bf16)f[2*i].z;   b[3]=(__bf16)f[2*i].w;
    b[4]=(__bf16)f[2*i+1].x; b[5]=(__bf16)f[2*i+1].y;
    b[6]=(__bf16)f[2*i+1].z; b[7]=(__bf16)f[2*i+1].w;
    *(bf16x8*)(d + 8 * i) = b;
  }
}

// ---- gfx1250 async copy: 16B global -> LDS per lane (ASYNCcnt tracked) ----
__device__ inline void async_copy_b128(unsigned lds_off, const void* gptr) {
  asm volatile("global_load_async_to_lds_b128 %0, %1, off"
               :: "v"(lds_off), "v"(gptr)
               : "memory");
}
__device__ inline void wait_asynccnt0() {
  asm volatile("s_wait_asynccnt 0" ::: "memory");
}

// 128 rows x 32 cols bf16 tile staged asynchronously: 2 x b128 per thread.
__device__ inline void stage_bf16_tile_async(__bf16* dst, const __bf16* src,
                                             int ld, int tid) {
  const int row = tid >> 1;
  const int col = (tid & 1) * 16;
  const __bf16* g = src + (size_t)row * ld + col;
  // Generic LDS pointer: low 32 bits are the LDS byte address (flat aperture).
  unsigned l = (unsigned)(uintptr_t)(dst + row * BKP + col);
  async_copy_b128(l,      g);
  async_copy_b128(l + 16, g + 8);
}

// ---------------------------------------------------------------------------
// Kernel 1: fused  h' = silu(x w1^T) * (x w3^T)  -> bf16 hbuf [T, HD]
// grid: (HD/BN=22, TPE/BM=8, E=8), block 256 = 8 waves as 2(M) x 4(N),
// wave tile 64x64 -> 4x4 C tiles per branch, 32 WMMA per K-slice per wave.
// ---------------------------------------------------------------------------
__global__ __launch_bounds__(256)
void moe_swiglu_gemm1(const float* __restrict__ x,
                      const float* __restrict__ w1,
                      const float* __restrict__ w3,
                      __bf16* __restrict__ hout) {
  __shared__ __bf16 As [2][BM * BKP];   // 2 x 10 KB
  __shared__ __bf16 B1s[2][BN * BKP];   // 2 x 20 KB
  __shared__ __bf16 B3s[2][BN * BKP];   // 2 x 20 KB

  const int nb   = blockIdx.x;
  const int mb   = blockIdx.y;
  const int e    = blockIdx.z;
  const int tid  = threadIdx.x;
  const int lane = tid & 31;
  const int wave = tid >> 5;
  const int wm   = wave & 1;                // 2 waves along M -> 64 rows
  const int wn   = wave >> 1;               // 4 waves along N -> 64 cols

  const float* xg  = x  + (size_t)e * TPE * D + (size_t)mb * BM * D;
  const float* w1g = w1 + (size_t)e * HD  * D + (size_t)nb * BN * D;
  const float* w3g = w3 + (size_t)e * HD  * D + (size_t)nb * BN * D;

  v8f acc1[4][4] = {};
  v8f acc3[4][4] = {};

  // Prologue: stage slice 0 into buffer 0.
  stage_f32_tile128(As[0],  xg,  D, tid);
  stage_f32_tile256(B1s[0], w1g, D, tid);
  stage_f32_tile256(B3s[0], w3g, D, tid);
  __syncthreads();

  int cur = 0;
  for (int k0 = 0; k0 < D; k0 += BK) {
    // Stage next slice into the other buffer (no conflict with cur reads).
    if (k0 + BK < D) {
      stage_f32_tile128(As[cur ^ 1],  xg  + k0 + BK, D, tid);
      stage_f32_tile256(B1s[cur ^ 1], w1g + k0 + BK, D, tid);
      stage_f32_tile256(B3s[cur ^ 1], w3g + k0 + BK, D, tid);
      if (k0 + 2 * BK < D)                  // global_prefetch_b8, 2 ahead
        __builtin_prefetch(xg + (size_t)(tid >> 1) * D + k0 + 2 * BK, 0, 3);
    }

    v16bf a[4], b1[4], b3[4];
#pragma unroll
    for (int i = 0; i < 4; ++i)
      a[i] = load_frag_A(&As[cur][(wm * 64 + i * 16) * BKP], lane);
#pragma unroll
    for (int j = 0; j < 4; ++j) {
      b1[j] = load_frag_B(&B1s[cur][(wn * 64 + j * 16) * BKP], lane);
      b3[j] = load_frag_B(&B3s[cur][(wn * 64 + j * 16) * BKP], lane);
    }
#pragma unroll
    for (int i = 0; i < 4; ++i)
#pragma unroll
      for (int j = 0; j < 4; ++j) {
        acc1[i][j] = __builtin_amdgcn_wmma_f32_16x16x32_bf16(
            false, a[i], false, b1[j], (short)0, acc1[i][j], false, false);
        acc3[i][j] = __builtin_amdgcn_wmma_f32_16x16x32_bf16(
            false, a[i], false, b3[j], (short)0, acc3[i][j], false, false);
      }

    __syncthreads();   // publishes next stage; retires cur reads (DScnt wait)
    cur ^= 1;
  }

  // Epilogue: SwiGLU gate in registers, bf16 store.
  // C tile layout: VGPR r, lane L: N = L&15, M = r + (L>=16 ? 8 : 0).
  const int cn  = lane & 15;
  const int mhi = (lane >> 4) * 8;
#pragma unroll
  for (int i = 0; i < 4; ++i)
#pragma unroll
    for (int j = 0; j < 4; ++j) {
      const int row0 = mb * BM + wm * 64 + i * 16 + mhi;
      const int col  = nb * BN + wn * 64 + j * 16 + cn;
      __bf16* o = hout + ((size_t)e * TPE + row0) * HD + col;
      v8f hv = acc1[i][j];
      v8f gv = acc3[i][j];
#pragma unroll
      for (int r = 0; r < 8; ++r) {
        float h = hv[r];
        float s = (h / (1.0f + __expf(-h))) * gv[r];   // silu(h) * g
        o[(size_t)r * HD] = (__bf16)s;
      }
    }
}

// ---------------------------------------------------------------------------
// Kernel 2: out = h' w2^T   -> fp32 [T, D]
// grid: (D/BN=8, TPE/BM=8, E=8), block 256, wave tile 64x64 (16 WMMA/slice).
// A (bf16) staged with gfx1250 async global->LDS copies.
// ---------------------------------------------------------------------------
__global__ __launch_bounds__(256)
void moe_gemm2(const __bf16* __restrict__ hbuf,
               const float* __restrict__ w2,
               float* __restrict__ out) {
  __shared__ __bf16 As[2][BM * BKP];    // 2 x 10 KB, async-filled
  __shared__ __bf16 Bs[2][BN * BKP];    // 2 x 20 KB

  const int nb   = blockIdx.x;
  const int mb   = blockIdx.y;
  const int e    = blockIdx.z;
  const int tid  = threadIdx.x;
  const int lane = tid & 31;
  const int wave = tid >> 5;
  const int wm   = wave & 1;
  const int wn   = wave >> 1;

  const __bf16* ag = hbuf + ((size_t)e * TPE + (size_t)mb * BM) * HD;
  const float*  bg = w2   + (size_t)e * D * HD + (size_t)nb * BN * HD;

  v8f acc[4][4] = {};

  stage_bf16_tile_async(As[0], ag, HD, tid);
  stage_f32_tile256(Bs[0], bg, HD, tid);
  wait_asynccnt0();
  __syncthreads();

  int cur = 0;
  for (int k0 = 0; k0 < HD; k0 += BK) {
    if (k0 + BK < HD) {
      stage_bf16_tile_async(As[cur ^ 1], ag + k0 + BK, HD, tid);
      stage_f32_tile256(Bs[cur ^ 1], bg + k0 + BK, HD, tid);
      if (k0 + 2 * BK < HD)
        __builtin_prefetch(bg + (size_t)tid * HD + k0 + 2 * BK, 0, 3);
    }

    v16bf a[4], b[4];
#pragma unroll
    for (int i = 0; i < 4; ++i)
      a[i] = load_frag_A(&As[cur][(wm * 64 + i * 16) * BKP], lane);
#pragma unroll
    for (int j = 0; j < 4; ++j)
      b[j] = load_frag_B(&Bs[cur][(wn * 64 + j * 16) * BKP], lane);
#pragma unroll
    for (int i = 0; i < 4; ++i)
#pragma unroll
      for (int j = 0; j < 4; ++j)
        acc[i][j] = __builtin_amdgcn_wmma_f32_16x16x32_bf16(
            false, a[i], false, b[j], (short)0, acc[i][j], false, false);

    wait_asynccnt0();  // async A stage retired (own ops) before publish
    __syncthreads();
    cur ^= 1;
  }

  const int cn  = lane & 15;
  const int mhi = (lane >> 4) * 8;
#pragma unroll
  for (int i = 0; i < 4; ++i)
#pragma unroll
    for (int j = 0; j < 4; ++j) {
      const int row0 = mb * BM + wm * 64 + i * 16 + mhi;
      const int col  = nb * BN + wn * 64 + j * 16 + cn;
      float* o = out + ((size_t)e * TPE + row0) * D + col;
      v8f c = acc[i][j];
#pragma unroll
      for (int r = 0; r < 8; ++r)
        o[(size_t)r * D] = c[r];
    }
}

// ---------------------------------------------------------------------------
extern "C" void kernel_launch(void* const* d_in, const int* in_sizes, int n_in,
                              void* d_out, int out_size, void* d_ws,
                              size_t ws_size, hipStream_t stream) {
  (void)in_sizes; (void)n_in; (void)out_size; (void)ws_size;
  const float* inp = (const float*)d_in[0];
  // d_in[1] = fwd_expert_count: equal counts (T/E) by construction -> unused.
  const float* w1  = (const float*)d_in[2];
  const float* w3  = (const float*)d_in[3];
  const float* w2  = (const float*)d_in[4];
  float*  out  = (float*)d_out;
  __bf16* hbuf = (__bf16*)d_ws;            // T*HD bf16 = 92.3 MB scratch

  dim3 g1(HD / BN, TPE / BM, E);           // (22, 8, 8)
  moe_swiglu_gemm1<<<g1, 256, 0, stream>>>(inp, w1, w3, hbuf);

  dim3 g2(D / BN, TPE / BM, E);            // (8, 8, 8)
  moe_gemm2<<<g2, 256, 0, stream>>>(hbuf, w2, out);
}
